// Cy2Mixer_layer_83915071029395
// MI455X (gfx1250) — matrix operand order repaired and checked by reading, not compiled, not run on hardware
//
#include <hip/hip_runtime.h>
#include <cstdint>
#include <cstddef>

// ---------------------------------------------------------------------------
// Types
// ---------------------------------------------------------------------------
typedef __bf16 bf16_t;
typedef __bf16 v16bf __attribute__((ext_vector_type(16)));
typedef __bf16 v8bf  __attribute__((ext_vector_type(8)));
typedef float  v8f   __attribute__((ext_vector_type(8)));

// Pointer types matching __builtin_amdgcn_global_load_async_to_lds_b128:
//   param0: AS1 (global)  v4i*  (vector_size(16) int, non-const)
//   param1: AS3 (LDS)     v4i*
typedef int v4i_vs __attribute__((__vector_size__(16)));
typedef __attribute__((address_space(1))) v4i_vs* gl_v4i_p;
typedef __attribute__((address_space(3))) v4i_vs* ld_v4i_p;
typedef __attribute__((address_space(3))) bf16_t* lds_bf16_p;

#define DEV_ __device__ __forceinline__

// Problem constants (match reference)
constexpr int CB = 4, CT = 64, CN = 128, CD = 256, CF = 512;
constexpr int TOK = CB * CT * CN;        // 32768 tokens
constexpr float LN_EPS = 1e-5f;

// GEMM tiling
constexpr int BM = 128, BN = 128, BK = 32;
constexpr int LDSS = BK + 16;            // 48 elems: pad keeps 16B/32B alignment

DEV_ v8f wmma_bf16(v16bf a, v16bf b, v8f c) {
  return __builtin_amdgcn_wmma_f32_16x16x32_bf16(
      /*neg_a=*/false, a, /*neg_b=*/false, b,
      /*c_mod=*/(short)0, c, /*reuse_a=*/false, /*reuse_b=*/false);
}

DEV_ float gelu_exact(float v) {
  return 0.5f * v * (1.0f + erff(v * 0.70710678118654752f));
}

// ---- CDNA5 async global->LDS copy (ASYNCcnt) -------------------------------
DEV_ void async_copy16(const bf16_t* g, bf16_t* l) {
#if __has_builtin(__builtin_amdgcn_global_load_async_to_lds_b128)
  __builtin_amdgcn_global_load_async_to_lds_b128(
      (gl_v4i_p)g, (ld_v4i_p)l, 0, 0);
#else
  asm volatile("global_load_async_to_lds_b128 %0, %1, off"
               :: "v"((lds_bf16_p)l), "v"(g)
               : "memory");
#endif
}

DEV_ void wait_async0() {
#if __has_builtin(__builtin_amdgcn_s_wait_asynccnt)
  __builtin_amdgcn_s_wait_asynccnt(0);
#else
  asm volatile("s_wait_asynccnt 0x0" ::: "memory");
#endif
}

// ---------------------------------------------------------------------------
// Generic WMMA GEMM:  C[M,N] = A[M,K](bf16,row-major,lda) x B + epilogue
//   B_IS_NK=true : B is [N,K] row-major (PyTorch Linear weight) with ldb
//   B_IS_NK=false: B is [K,N] row-major with ldb (transposed while staging)
// Block = 256 threads (8 waves), block tile 128x128, k-step 32.
// Wave grid 2(m) x 4(n); per wave 4x2 tiles of 16x16 -> 8 v8f accumulators.
// Double-buffered LDS; A (and B for NK) staged with async-to-LDS copies.
// ---------------------------------------------------------------------------
template <bool B_IS_NK, bool OUT_BF16, bool HAS_BIAS, bool DO_GELU,
          bool HAS_RES, bool ACCUM>
__global__ __launch_bounds__(256) void wmma_gemm(
    const bf16_t* __restrict__ A, const bf16_t* __restrict__ Bm,
    float* __restrict__ outF, bf16_t* __restrict__ outB,
    const float* __restrict__ bias, const float* __restrict__ res,
    int M, int N, int K, int lda, int ldb, int ldc,
    long long strideA, long long strideB, long long strideC) {
  __shared__ __attribute__((aligned(32))) bf16_t sA[2][BM * LDSS];
  __shared__ __attribute__((aligned(32))) bf16_t sB[2][BN * LDSS];

  const int tid = threadIdx.x;
  const int z = blockIdx.z;
  A  += (long long)z * strideA;
  Bm += (long long)z * strideB;
  const long long cbase = (long long)z * strideC;

  const int m0 = blockIdx.y * BM;
  const int n0 = blockIdx.x * BN;
  const int wid = tid >> 5, lane = tid & 31;
  const int wm = wid >> 2, wn = wid & 3;     // 2 x 4 waves
  const int half = lane >> 4, lq = lane & 15;

  v8f acc[4][2];
#pragma unroll
  for (int i = 0; i < 4; ++i)
#pragma unroll
    for (int j = 0; j < 2; ++j)
#pragma unroll
      for (int e = 0; e < 8; ++e) acc[i][j][e] = 0.0f;

  // Stage one BKxBM / BKxBN tile pair into LDS buffer `buf`.
  auto stage = [&](int kk, int buf) {
    {
      const int r = tid >> 1;
      const int c16 = (tid & 1) * 16;
      const bf16_t* gp = A + (size_t)(m0 + r) * lda + kk + c16;
      bf16_t* lp = &sA[buf][r * LDSS + c16];
      async_copy16(gp, lp);
      async_copy16(gp + 8, lp + 8);
    }
    if (B_IS_NK) {
      const int r = tid >> 1;
      const int c16 = (tid & 1) * 16;
      const bf16_t* gp = Bm + (size_t)(n0 + r) * ldb + kk + c16;
      bf16_t* lp = &sB[buf][r * LDSS + c16];
      async_copy16(gp, lp);
      async_copy16(gp + 8, lp + 8);
    } else {
#pragma unroll
      for (int i = 0; i < 2; ++i) {
        const int ch = tid * 2 + i;      // 0..511
        const int k  = ch >> 4;          // 0..31
        const int nc = (ch & 15) * 8;    // 0..120
        uint4 d = *(const uint4*)(Bm + (size_t)(kk + k) * ldb + n0 + nc);
        const bf16_t* e = (const bf16_t*)&d;
#pragma unroll
        for (int j = 0; j < 8; ++j) sB[buf][(nc + j) * LDSS + k] = e[j];
      }
    }
  };

  stage(0, 0);
  int buf = 0;
  for (int kk = 0; kk < K; kk += BK) {
    wait_async0();          // this wave's tile (and prior issues) landed in LDS
    __syncthreads();        // all waves' tiles visible; prior compute finished
    if (kk + BK < K) stage(kk + BK, buf ^ 1);

    // ---- fragments + WMMA on current buffer ----
    v16bf afrag[4], bfrag[2];
#pragma unroll
    for (int tm = 0; tm < 4; ++tm) {
      const int m = wm * 64 + tm * 16 + lq;
      const bf16_t* p = &sA[buf][m * LDSS + half * 16];
      v8bf lo = *(const v8bf*)p;
      v8bf hi = *(const v8bf*)(p + 8);
      afrag[tm] = __builtin_shufflevector(lo, hi, 0, 1, 2, 3, 4, 5, 6, 7,
                                          8, 9, 10, 11, 12, 13, 14, 15);
    }
#pragma unroll
    for (int tn = 0; tn < 2; ++tn) {
      const int n = wn * 32 + tn * 16 + lq;
      bfrag[tn] = *(const v16bf*)&sB[buf][n * LDSS + half * 16];
    }
#pragma unroll
    for (int tm = 0; tm < 4; ++tm)
#pragma unroll
      for (int tn = 0; tn < 2; ++tn)
        acc[tm][tn] = wmma_bf16(afrag[tm], bfrag[tn], acc[tm][tn]);
    buf ^= 1;
  }

  // ---- epilogue ----
#pragma unroll
  for (int tm = 0; tm < 4; ++tm)
#pragma unroll
    for (int tn = 0; tn < 2; ++tn)
#pragma unroll
      for (int e = 0; e < 8; ++e) {
        const int row = m0 + wm * 64 + tm * 16 + half * 8 + e;
        const int col = n0 + wn * 32 + tn * 16 + lq;
        float v = acc[tm][tn][e];
        if (HAS_BIAS) v += bias[col];
        if (DO_GELU) v = gelu_exact(v);
        if (HAS_RES) v += res[(size_t)row * ldc + col];
        const long long idx = cbase + (long long)row * ldc + col;
        if (ACCUM) v += outF[idx];
        if (OUT_BF16) outB[idx] = (bf16_t)v;
        else          outF[idx] = v;
      }
}

// ---------------------------------------------------------------------------
// Conv2d(T,T,(1,3),pad(0,1)) over feature axis as 3 shifted 64x64x512 WMMA
// GEMMs per (b,n) slice.  vn lives in-place in huv[:, F:2F] (stride 2F).
// Block: 256 thr, BM=64 (all t_o), BN=128 f; waves 4(m) x 2(n); per wave
// 1x4 tiles.  cw (3x64x64 bf16) async-staged in LDS; B frags from global.
// ---------------------------------------------------------------------------
__global__ __launch_bounds__(256) void conv_wmma(
    const bf16_t* __restrict__ cwb,   // [3][64][64] bf16 (dw, t_o, t_i)
    const bf16_t* __restrict__ huv,   // [TOK][2F], vn at +CF
    const float* __restrict__ cb,     // [T]
    bf16_t* __restrict__ gbuf) {      // [TOK][F]
  __shared__ __attribute__((aligned(16))) bf16_t sc[3 * 64 * 64];
  const int tid = threadIdx.x;
#pragma unroll
  for (int i = 0; i < 6; ++i) {
    const int c = i * 256 + tid;      // 1536 chunks of 8 elements
    async_copy16(cwb + c * 8, &sc[c * 8]);
  }
  wait_async0();
  __syncthreads();

  const int z = blockIdx.z;
  const int b = z >> 7, nn = z & 127;
  const int fblk = blockIdx.x * 128;
  const int wid = tid >> 5, lane = tid & 31;
  const int wm = wid >> 1, wn = wid & 1;   // 4 x 2 waves
  const int half = lane >> 4, lq = lane & 15;

  v8f acc[4];
#pragma unroll
  for (int tn = 0; tn < 4; ++tn)
#pragma unroll
    for (int e = 0; e < 8; ++e) acc[tn][e] = 0.0f;

  const size_t slice = ((size_t)(b * CT) * CN + nn) * (size_t)(2 * CF) + CF;
  const size_t rowstride = (size_t)CN * 2 * CF;  // step t_i -> next token row

  for (int dw = 0; dw < 3; ++dw) {
    for (int kkb = 0; kkb < 64; kkb += 32) {
      // A fragment from LDS cw[dw]
      const bf16_t* ap = sc + dw * 4096 + (wm * 16 + lq) * 64 + kkb + half * 16;
      v8bf lo = *(const v8bf*)ap;
      v8bf hi = *(const v8bf*)(ap + 8);
      v16bf af = __builtin_shufflevector(lo, hi, 0, 1, 2, 3, 4, 5, 6, 7,
                                         8, 9, 10, 11, 12, 13, 14, 15);
#pragma unroll
      for (int tn = 0; tn < 4; ++tn) {
        const int f = fblk + wn * 64 + tn * 16 + lq;
        const int fs = f + dw - 1;
        v16bf bf;
        if (fs >= 0 && fs < CF) {
          const bf16_t* vp = huv + slice + fs;
          const int kb = kkb + half * 16;
#pragma unroll
          for (int j = 0; j < 16; ++j)
            bf[j] = vp[(size_t)(kb + j) * rowstride];
        } else {
#pragma unroll
          for (int j = 0; j < 16; ++j) bf[j] = (bf16_t)0.0f;
        }
        acc[tn] = wmma_bf16(af, bf, acc[tn]);
      }
    }
  }
#pragma unroll
  for (int tn = 0; tn < 4; ++tn)
#pragma unroll
    for (int e = 0; e < 8; ++e) {
      const int m = wm * 16 + half * 8 + e;             // t_o
      const int f = fblk + wn * 64 + tn * 16 + lq;
      const float v = acc[tn][e] + cb[m];
      gbuf[((size_t)(b * CT + m) * CN + nn) * CF + f] = (bf16_t)v;
    }
}

// ---------------------------------------------------------------------------
// Elementwise / reduction helpers
// ---------------------------------------------------------------------------
DEV_ float block_reduce_sum(float v, float* red, int tid) {
  red[tid] = v;
  __syncthreads();
#pragma unroll
  for (int st = 128; st > 0; st >>= 1) {
    if (tid < st) red[tid] += red[tid + st];
    __syncthreads();
  }
  const float r = red[0];
  __syncthreads();
  return r;
}

// LN over last dim 256: fp32 in -> bf16 out
__global__ __launch_bounds__(256) void ln256_f32_bf16(
    const float* __restrict__ in, const float* __restrict__ g,
    const float* __restrict__ b, bf16_t* __restrict__ outp) {
  __shared__ float red[256];
  const int row = blockIdx.x, tid = threadIdx.x;
  const float v = in[(size_t)row * CD + tid];
  const float mean = block_reduce_sum(v, red, tid) * (1.0f / CD);
  const float d = v - mean;
  const float var = block_reduce_sum(d * d, red, tid) * (1.0f / CD);
  const float rs = rsqrtf(var + LN_EPS);
  outp[(size_t)row * CD + tid] = (bf16_t)(d * rs * g[tid] + b[tid]);
}

// LN over F=512 of v-half of huv (bf16), in place
__global__ __launch_bounds__(256) void ln_v_inplace(
    bf16_t* __restrict__ huv, const float* __restrict__ g,
    const float* __restrict__ b) {
  __shared__ float red[256];
  const int row = blockIdx.x, tid = threadIdx.x;
  bf16_t* p = huv + (size_t)row * (2 * CF) + CF;
  const float a0 = (float)p[tid], a1 = (float)p[tid + 256];
  const float mean = block_reduce_sum(a0 + a1, red, tid) * (1.0f / CF);
  const float d0 = a0 - mean, d1 = a1 - mean;
  const float var = block_reduce_sum(d0 * d0 + d1 * d1, red, tid) * (1.0f / CF);
  const float rs = rsqrtf(var + LN_EPS);
  p[tid]       = (bf16_t)(d0 * rs * g[tid] + b[tid]);
  p[tid + 256] = (bf16_t)(d1 * rs * g[tid + 256] + b[tid + 256]);
}

// p = u * g   (u = huv[:, :F])
__global__ __launch_bounds__(256) void gate_mul(
    const bf16_t* __restrict__ huv, const bf16_t* __restrict__ gbuf,
    bf16_t* __restrict__ pbuf) {
  const long long i = (long long)blockIdx.x * 256 + threadIdx.x;
  const long long tok = i >> 9;
  const int f = (int)(i & 511);
  const float u = (float)huv[tok * (2 * CF) + f];
  const float gv = (float)gbuf[i];
  pbuf[i] = (bf16_t)(u * gv);
}

// fp32 -> bf16
__global__ __launch_bounds__(256) void cvt_bf16(const float* __restrict__ s,
                                                bf16_t* __restrict__ d, int n) {
  const int i = blockIdx.x * 256 + threadIdx.x;
  if (i < n) d[i] = (bf16_t)s[i];
}

// cw [T,T,1,3] fp32 -> [3][64][64] bf16
__global__ __launch_bounds__(256) void cvt_cw(const float* __restrict__ s,
                                              bf16_t* __restrict__ d) {
  const int i = blockIdx.x * 256 + threadIdx.x;  // over 3*64*64
  if (i < 3 * 64 * 64) {
    const int dw = i / 4096, r = i - dw * 4096;
    const int to = r >> 6, ti = r & 63;
    d[i] = (bf16_t)s[(to * 64 + ti) * 3 + dw];
  }
}

// Normalized adjacency transposed: aT[m][k] = a_hat[k][m], bf16
__global__ __launch_bounds__(128) void adj_norm(const int* __restrict__ cir,
                                                bf16_t* __restrict__ aT) {
  __shared__ float dinv[CN];
  const int i = threadIdx.x;
  float deg = 0.0f;
  for (int r = 0; r < CN; ++r) {
    const float a = (r == i) ? 1.0f : (cir[r * CN + i] != 0 ? 1.0f : 0.0f);
    deg += a;
  }
  dinv[i] = rsqrtf(deg);
  __syncthreads();
  const float di = dinv[i];
  for (int k = 0; k < CN; ++k) {
    const float a = (k == i) ? 1.0f : (cir[k * CN + i] != 0 ? 1.0f : 0.0f);
    aT[i * CN + k] = (bf16_t)(a * dinv[k] * di);
  }
}

// ---------------------------------------------------------------------------
// Host-side orchestration
// ---------------------------------------------------------------------------
extern "C" void kernel_launch(void* const* d_in, const int* in_sizes, int n_in,
                              void* d_out, int out_size, void* d_ws,
                              size_t ws_size, hipStream_t stream) {
  (void)in_sizes; (void)n_in; (void)out_size; (void)ws_size;
  const float* x = (const float*)d_in[0];
  const int* cir = (const int*)d_in[1];
  auto P = [&](int i) { return (const float*)d_in[i]; };
  // inputs: 2..9 tgu_*, 10..17 sgu_*, 18..25 cgu_*  (ng,nb,w1,b1,sg,sb,w2,b2)
  const float* cw    = P(26);
  const float* cb    = P(27);
  const float* gw_s  = P(28);
  const float* gb_s  = P(29);
  const float* gw_c  = P(30);
  const float* gb_c  = P(31);
  const float* n_g[3] = {P(32), P(34), P(36)};
  const float* n_b[3] = {P(33), P(35), P(37)};
  const float* w_out = P(38);
  const float* b_out = P(39);
  float* out = (float*)d_out;

  // ---- workspace carve-out ----
  char* ws = (char*)d_ws;
  size_t off = 0;
  auto carve = [&](size_t bytes) {
    void* p = ws + off;
    off += (bytes + 255) & ~(size_t)255;
    return p;
  };
  bf16_t* aT    = (bf16_t*)carve((size_t)CN * CN * 2);
  bf16_t* w1b   = (bf16_t*)carve((size_t)3 * 2 * CF * CD * 2);
  bf16_t* gwb   = (bf16_t*)carve((size_t)2 * CF * CF * 2);
  bf16_t* w2b   = (bf16_t*)carve((size_t)3 * CD * CF * 2);
  bf16_t* woutb = (bf16_t*)carve((size_t)CD * 3 * CD * 2);
  bf16_t* cwb   = (bf16_t*)carve((size_t)3 * CT * CT * 2);
  bf16_t* lnx   = (bf16_t*)carve((size_t)TOK * CD * 2);
  bf16_t* huv   = (bf16_t*)carve((size_t)TOK * 2 * CF * 2);
  bf16_t* ybuf  = (bf16_t*)carve((size_t)TOK * CF * 2);
  bf16_t* gbuf  = (bf16_t*)carve((size_t)TOK * CF * 2);
  float*  bo    = (float*)carve((size_t)TOK * CD * 4);

  // ---- one-time (per launch) prep ----
  adj_norm<<<1, 128, 0, stream>>>(cir, aT);
  auto cvt = [&](const float* s, bf16_t* d, int n) {
    cvt_bf16<<<(n + 255) / 256, 256, 0, stream>>>(s, d, n);
  };
  for (int br = 0; br < 3; ++br) {
    const int base = 2 + br * 8;
    cvt(P(base + 2), w1b + (size_t)br * 2 * CF * CD, 2 * CF * CD);
    cvt(P(base + 6), w2b + (size_t)br * CD * CF, CD * CF);
  }
  cvt(gw_s, gwb, CF * CF);
  cvt(gw_c, gwb + (size_t)CF * CF, CF * CF);
  cvt(w_out, woutb, CD * 3 * CD);
  cvt_cw<<<(3 * CT * CT + 255) / 256, 256, 0, stream>>>(cw, cwb);

  // ---- three branches: 0=temporal(conv), 1=spatial(GCN), 2=cycle(GCN) ----
  for (int br = 0; br < 3; ++br) {
    const int base = 2 + br * 8;
    const float* ng = P(base + 0);
    const float* nb = P(base + 1);
    const float* b1 = P(base + 3);
    const float* sg = P(base + 4);
    const float* sb = P(base + 5);
    const float* b2 = P(base + 7);

    // lnx = LN(x; ng, nb)  (bf16)
    ln256_f32_bf16<<<TOK, 256, 0, stream>>>(x, ng, nb, lnx);

    // huv = gelu(lnx @ w1^T + b1)   [TOK, 2F] bf16
    wmma_gemm<true, true, true, true, false, false>
        <<<dim3(2 * CF / BN, TOK / BM, 1), 256, 0, stream>>>(
            lnx, w1b + (size_t)br * 2 * CF * CD, nullptr, huv, b1, nullptr,
            TOK, 2 * CF, CD, CD, CD, 2 * CF, 0, 0, 0);

    // vn = LN(v; sg, sb) in place
    ln_v_inplace<<<TOK, 256, 0, stream>>>(huv, sg, sb);

    if (br == 0) {
      // temporal: conv mixer -> gbuf (+cb)
      conv_wmma<<<dim3(CF / BN, 1, CB * CN), 256, 0, stream>>>(cwb, huv, cb,
                                                               gbuf);
    } else {
      const float* gb = (br == 1) ? gb_s : gb_c;
      bf16_t* gwp = gwb + (size_t)(br - 1) * CF * CF;
      // ybuf = vn @ gw^T   [TOK, F]
      wmma_gemm<true, true, false, false, false, false>
          <<<dim3(CF / BN, TOK / BM, 1), 256, 0, stream>>>(
              huv + CF, gwp, nullptr, ybuf, nullptr, nullptr,
              TOK, CF, CF, 2 * CF, CF, CF, 0, 0, 0);
      // gbuf = a_hat^T x ybuf_slice + gb   (batched over B*T slices)
      wmma_gemm<false, true, true, false, false, false>
          <<<dim3(CF / BN, 1, CB * CT), 256, 0, stream>>>(
              aT, ybuf, nullptr, gbuf, gb, nullptr,
              CN, CF, CN, CN, CF, CF, 0, (long long)CN * CF,
              (long long)CN * CF);
    }

    // ybuf = u * gbuf
    gate_mul<<<(TOK * CF) / 256, 256, 0, stream>>>(huv, gbuf, ybuf);

    // bo = ybuf @ w2^T + b2 + x   (fp32)
    wmma_gemm<true, false, true, false, true, false>
        <<<dim3(CD / BN, TOK / BM, 1), 256, 0, stream>>>(
            ybuf, w2b + (size_t)br * CD * CF, bo, nullptr, b2, x,
            TOK, CD, CF, CF, CF, CD, 0, 0, 0);

    // lnx = LN(bo; n{1,2,3})
    ln256_f32_bf16<<<TOK, 256, 0, stream>>>(bo, n_g[br], n_b[br], lnx);

    // out (+)= lnx @ w_out[:, br*D:(br+1)*D]^T ; branch0 adds b_out + x
    if (br == 0) {
      wmma_gemm<true, false, true, false, true, false>
          <<<dim3(CD / BN, TOK / BM, 1), 256, 0, stream>>>(
              lnx, woutb, out, nullptr, b_out, x,
              TOK, CD, CD, CD, 3 * CD, CD, 0, 0, 0);
    } else {
      wmma_gemm<true, false, false, false, false, true>
          <<<dim3(CD / BN, TOK / BM, 1), 256, 0, stream>>>(
              lnx, woutb + (size_t)br * CD, out, nullptr, nullptr, nullptr,
              TOK, CD, CD, CD, 3 * CD, CD, 0, 0, 0);
    }
  }
}
